// BruteForceMoE_60679297957912
// MI455X (gfx1250) — matrix operand order repaired and checked by reading, not compiled, not run on hardware
//
#include <hip/hip_runtime.h>

// MoE brute-force forward for gfx1250 (MI455X), bf16x3 split-precision WMMA,
// register-double-buffered global->LDS staging.
// inputs: inp[2048,1024] f32, gate_idx[2048,2] i32, gate_score[2048,2] f32,
//         W[16,1024,1024] f32, b[16,1024] f32 -> y[2048,1024] f32

#define NUM_EXPERT 16
#define D_MODEL    1024
#define TOP_K      2
#define N_TOKENS   2048
#define NSLOT      (N_TOKENS * TOP_K)   // 4096 dispatched (token,slot) pairs
#define TM         64                    // rows per block (bucketed slots)
#define TN         64                    // output cols per block
#define KC         32                    // K chunk (bf16 WMMA K)
#define LDSTR      40                    // padded LDS stride (elems; 80B rows: 16B-aligned, bank-spread)
#define MAXMT      (NSLOT / TM)          // 64 worst-case row tiles per expert

typedef __bf16 bfv16 __attribute__((ext_vector_type(16)));
typedef __bf16 bfv8  __attribute__((ext_vector_type(8)));
typedef float  fv8   __attribute__((ext_vector_type(8)));

static __device__ __forceinline__ bfv16 load_frag(const __bf16* p, const __bf16* q) {
  bfv8 a = *(const bfv8*)p;   // 16B aligned
  bfv8 b = *(const bfv8*)q;
  return __builtin_shufflevector(a, b, 0,1,2,3,4,5,6,7,8,9,10,11,12,13,14,15);
}

static __device__ __forceinline__ fv8 wmma_bf16(bfv16 a, bfv16 b, fv8 c) {
  return __builtin_amdgcn_wmma_f32_16x16x32_bf16(false, a, false, b, (short)0, c, false, false);
}

// split f32 -> (hi, lo) bf16 and store 4 consecutive K elements of a row
static __device__ __forceinline__ void cvt_store_row(__bf16* hi, __bf16* lo, int idx, float4 v) {
  float a[4] = {v.x, v.y, v.z, v.w};
  #pragma unroll
  for (int j = 0; j < 4; ++j) {
    __bf16 h = (__bf16)a[j];
    hi[idx + j] = h;
    lo[idx + j] = (__bf16)(a[j] - (float)h);
  }
}

// split f32 -> (hi, lo) bf16 and store 4 N elements transposed into [N][K] tile
static __device__ __forceinline__ void cvt_store_colT(__bf16* hi, __bf16* lo, int bn, int bk, float4 v) {
  float a[4] = {v.x, v.y, v.z, v.w};
  #pragma unroll
  for (int j = 0; j < 4; ++j) {
    __bf16 h = (__bf16)a[j];
    hi[(bn + j) * LDSTR + bk] = h;
    lo[(bn + j) * LDSTR + bk] = (__bf16)(a[j] - (float)h);
  }
}

// ---------------- bucket slots by expert (single block) ----------------
__global__ __launch_bounds__(256)
void moe_bucket(const int* __restrict__ gidx, int* __restrict__ offs, int* __restrict__ rows) {
  __shared__ int cnt[NUM_EXPERT];
  __shared__ int cur[NUM_EXPERT];
  const int tid = threadIdx.x;
  if (tid < NUM_EXPERT) cnt[tid] = 0;
  __syncthreads();
  for (int p = tid; p < NSLOT; p += 256) atomicAdd(&cnt[gidx[p]], 1);
  __syncthreads();
  if (tid == 0) {
    int run = 0;
    for (int e = 0; e < NUM_EXPERT; ++e) { offs[e] = run; cur[e] = run; run += cnt[e]; }
    offs[NUM_EXPERT] = run;
  }
  __syncthreads();
  for (int p = tid; p < NSLOT; p += 256) {
    int e = gidx[p];
    int pos = atomicAdd(&cur[e], 1);
    rows[pos] = p;   // per-slot results independent of bucket order -> deterministic output
  }
}

// ---------------- grouped GEMM: o[slot] = gs[slot] * (x[t] @ W[e] + b[e]) ----------------
__global__ __launch_bounds__(256)
void moe_gemm(const float* __restrict__ inp, const float* __restrict__ gs,
              const float* __restrict__ W, const float* __restrict__ bias,
              const int* __restrict__ offs, const int* __restrict__ rows,
              float* __restrict__ o)
{
  const int e    = blockIdx.x / MAXMT;
  const int mt   = blockIdx.x % MAXMT;
  const int n0   = blockIdx.y * TN;
  const int base = offs[e];
  const int ne   = offs[e + 1] - base;
  if (mt * TM >= ne) return;   // block-uniform exit (EXEC stays all-1 for WMMA)

  __shared__ __align__(16) __bf16 sAhi[TM * LDSTR], sAlo[TM * LDSTR];
  __shared__ __align__(16) __bf16 sBhi[TN * LDSTR], sBlo[TN * LDSTR];

  const int tid = threadIdx.x;

  // A loader: two rows/thread, float4 of K per row. Out-of-range rows are CLAMPED
  // to ne-1 (not zeroed): WMMA D row m depends only on A row m, and the epilogue
  // guards writes, so padding-row garbage is harmless. Loads stay unconditional.
  const int ar0 = tid >> 3;            // 0..31
  const int ar1 = ar0 + 32;            // 32..63
  const int ac  = (tid & 7) * 4;       // K offset within chunk
  int g0 = mt * TM + ar0; if (g0 >= ne) g0 = ne - 1;
  int g1 = mt * TM + ar1; if (g1 >= ne) g1 = ne - 1;
  const int rA0 = rows[base + g0];
  const int rA1 = rows[base + g1];
  const float* pxA0 = inp + (size_t)(rA0 >> 1) * D_MODEL + ac;
  const float* pxA1 = inp + (size_t)(rA1 >> 1) * D_MODEL + ac;

  // B loader: two K-rows/thread, float4 of N (coalesced along N)
  const int bk0 = tid >> 4;            // 0..15
  const int bk1 = bk0 + 16;            // 16..31
  const int bn  = (tid & 15) * 4;      // N offset within tile
  const float* Wp  = W + (size_t)e * D_MODEL * D_MODEL + n0;
  const float* pW0 = Wp + (size_t)bk0 * D_MODEL + bn;
  const float* pW1 = Wp + (size_t)bk1 * D_MODEL + bn;

  // wave/tile mapping: 8 waves; wave owns tiles (rb0,cb),(rb1,cb); B frag shared
  const int lane = tid & 31;
  const int w    = tid >> 5;
  const int cb   = w & 3;
  const int rb0  = w >> 2;             // 0..1
  const int rb1  = rb0 + 2;            // 2..3
  const int half = lane >> 4;
  const int lan  = lane & 15;

  fv8 acc0 = {0.f, 0.f, 0.f, 0.f, 0.f, 0.f, 0.f, 0.f};
  fv8 acc1 = acc0;

  // register double-buffer: prefetch chunk 0
  float4 rAv0 = *(const float4*)(pxA0);
  float4 rAv1 = *(const float4*)(pxA1);
  float4 rBv0 = *(const float4*)(pW0);
  float4 rBv1 = *(const float4*)(pW1);

  for (int kc = 0; kc < D_MODEL; kc += KC) {
    // ---- convert current chunk to hi/lo bf16 and stage in LDS ----
    cvt_store_row (sAhi, sAlo, ar0 * LDSTR + ac, rAv0);
    cvt_store_row (sAhi, sAlo, ar1 * LDSTR + ac, rAv1);
    cvt_store_colT(sBhi, sBlo, bn, bk0, rBv0);
    cvt_store_colT(sBhi, sBlo, bn, bk1, rBv1);
    __syncthreads();

    // ---- prefetch next chunk's f32 tiles (overlaps with WMMAs below) ----
    const int kn = kc + KC;
    if (kn < D_MODEL) {
      rAv0 = *(const float4*)(pxA0 + kn);
      rAv1 = *(const float4*)(pxA1 + kn);
      rBv0 = *(const float4*)(pW0 + (size_t)kn * D_MODEL);
      rBv1 = *(const float4*)(pW1 + (size_t)kn * D_MODEL);
    }

    // ---- compute: bf16x3 split-precision WMMA ----
    const __bf16* bh = sBhi + (cb * 16 + lan) * LDSTR + half * 8;
    const __bf16* bl = sBlo + (cb * 16 + lan) * LDSTR + half * 8;
    bfv16 Bh = load_frag(bh, bh + 16);
    bfv16 Bl = load_frag(bl, bl + 16);

    const __bf16* a0h = sAhi + (rb0 * 16 + lan) * LDSTR + half * 8;
    const __bf16* a0l = sAlo + (rb0 * 16 + lan) * LDSTR + half * 8;
    bfv16 A0h = load_frag(a0h, a0h + 16);
    bfv16 A0l = load_frag(a0l, a0l + 16);
    const __bf16* a1h = sAhi + (rb1 * 16 + lan) * LDSTR + half * 8;
    const __bf16* a1l = sAlo + (rb1 * 16 + lan) * LDSTR + half * 8;
    bfv16 A1h = load_frag(a1h, a1h + 16);
    bfv16 A1l = load_frag(a1l, a1l + 16);

    acc0 = wmma_bf16(A0h, Bh, acc0);
    acc0 = wmma_bf16(A0h, Bl, acc0);
    acc0 = wmma_bf16(A0l, Bh, acc0);
    acc1 = wmma_bf16(A1h, Bh, acc1);
    acc1 = wmma_bf16(A1h, Bl, acc1);
    acc1 = wmma_bf16(A1l, Bh, acc1);
    __syncthreads();
  }

  // ---- epilogue: o[slot, col] = (acc + bias) * gate_score[slot] ----
  const int col = n0 + cb * 16 + lan;
  const float bv = bias[e * D_MODEL + col];
  #pragma unroll
  for (int r = 0; r < 8; ++r) {
    int gRow = mt * TM + rb0 * 16 + half * 8 + r;   // C/D layout: VGPR r -> M = r + half*8
    if (gRow < ne) {
      int p = rows[base + gRow];
      o[(size_t)p * D_MODEL + col] = (acc0[r] + bv) * gs[p];
    }
  }
  #pragma unroll
  for (int r = 0; r < 8; ++r) {
    int gRow = mt * TM + rb1 * 16 + half * 8 + r;
    if (gRow < ne) {
      int p = rows[base + gRow];
      o[(size_t)p * D_MODEL + col] = (acc1[r] + bv) * gs[p];
    }
  }
}

// ---------------- combine the two slots per token ----------------
__global__ __launch_bounds__(256)
void moe_combine(const float* __restrict__ o, float* __restrict__ y) {
  const int i = blockIdx.x * 256 + threadIdx.x;       // float4 index over y (t*256 + c4)
  const int t = i >> 8;
  const float4* o4 = (const float4*)o;
  float4 a = o4[(size_t)i + (size_t)t * 256];         // slot 2t
  float4 b = o4[(size_t)i + (size_t)t * 256 + 256];   // slot 2t+1
  float4 r;
  r.x = a.x + b.x; r.y = a.y + b.y; r.z = a.z + b.z; r.w = a.w + b.w;
  ((float4*)y)[i] = r;
}

extern "C" void kernel_launch(void* const* d_in, const int* in_sizes, int n_in,
                              void* d_out, int out_size, void* d_ws, size_t ws_size,
                              hipStream_t stream) {
  (void)in_sizes; (void)n_in; (void)out_size; (void)ws_size;
  const float* inp  = (const float*)d_in[0];
  const int*   gidx = (const int*)  d_in[1];
  const float* gs   = (const float*)d_in[2];
  const float* W    = (const float*)d_in[3];
  const float* bias = (const float*)d_in[4];
  float* y = (float*)d_out;

  // workspace: [offs:32 i32][rows:4096 i32][o: 4096*1024 f32 = 16 MB], 16B aligned
  int*   offs = (int*)d_ws;
  int*   rows = offs + 32;
  float* o    = (float*)d_ws + 32 + NSLOT;

  moe_bucket<<<1, 256, 0, stream>>>(gidx, offs, rows);
  dim3 grid(NUM_EXPERT * MAXMT, D_MODEL / TN);
  moe_gemm<<<grid, 256, 0, stream>>>(inp, gs, W, bias, offs, rows, o);
  moe_combine<<<(N_TOKENS * D_MODEL / 4) / 256, 256, 0, stream>>>(o, y);
}
